// InfoNCE_34754875359471
// MI455X (gfx1250) — matrix operand order repaired and checked by reading, not compiled, not run on hardware
//
#include <hip/hip_runtime.h>

typedef _Float16 v16h __attribute__((ext_vector_type(16)));
typedef _Float16 v8h  __attribute__((ext_vector_type(8)));
typedef float    v8f  __attribute__((ext_vector_type(8)));

#define DDIM     256
#define KSTEP    8                      // 256 / 32
#define INV_T    14.285714285714286f    // 1 / 0.07
#define EXPSCL   (1.4426950408889634f * 14.285714285714286f)  // log2(e)/T
#define LDSROW   280                    // padded halves per tile row (bank-conflict-free, 16B aligned)
#define TILEB    (16 * LDSROW * 2)      // bytes per LDS tile buffer (8960)

// ---------------------------------------------------------------------------
// Kernel 1: L2-normalize rows of q and k -> f16, and compute diagonal logits.
// One wave32 per row; 8 elements per lane.
// ---------------------------------------------------------------------------
__global__ __launch_bounds__(256) void nrm_diag(const float* __restrict__ q,
                                                const float* __restrict__ k,
                                                _Float16* __restrict__ qh,
                                                _Float16* __restrict__ kh,
                                                float* __restrict__ diag) {
    const int wave = threadIdx.x >> 5;
    const int ln   = threadIdx.x & 31;
    const int r    = blockIdx.x * 8 + wave;

    const float4* qv = (const float4*)(q + (size_t)r * DDIM + ln * 8);
    const float4* kv = (const float4*)(k + (size_t)r * DDIM + ln * 8);
    float4 q0 = qv[0], q1 = qv[1];
    float4 k0 = kv[0], k1 = kv[1];

    float qa[8] = {q0.x, q0.y, q0.z, q0.w, q1.x, q1.y, q1.z, q1.w};
    float ka[8] = {k0.x, k0.y, k0.z, k0.w, k1.x, k1.y, k1.z, k1.w};

    float sq = 0.0f, sk = 0.0f, dd = 0.0f;
#pragma unroll
    for (int i = 0; i < 8; ++i) {
        sq += qa[i] * qa[i];
        sk += ka[i] * ka[i];
        dd += qa[i] * ka[i];
    }
#pragma unroll
    for (int m = 16; m > 0; m >>= 1) {
        sq += __shfl_xor(sq, m, 32);
        sk += __shfl_xor(sk, m, 32);
        dd += __shfl_xor(dd, m, 32);
    }
    const float invq = 1.0f / fmaxf(sqrtf(sq), 1e-12f);
    const float invk = 1.0f / fmaxf(sqrtf(sk), 1e-12f);

    v8h qo, ko;
#pragma unroll
    for (int i = 0; i < 8; ++i) {
        qo[i] = (_Float16)(qa[i] * invq);
        ko[i] = (_Float16)(ka[i] * invk);
    }
    *(v8h*)(qh + (size_t)r * DDIM + ln * 8) = qo;
    *(v8h*)(kh + (size_t)r * DDIM + ln * 8) = ko;

    if (ln == 0) diag[r] = dd * invq * invk * INV_T;
}

// ---------------------------------------------------------------------------
// Kernel 2: fused WMMA GEMM + logsumexp, async global->LDS triple buffering.
// 4 waves x 32 rows = 128 rows per block. TWO 16x256 A panels per wave
// resident in VGPRs (16 fragments, 128 VGPRs) so every B fragment read from
// LDS feeds two independent WMMAs -> half the ds traffic / waits / barriers
// per FLOP. k tiles (16 cols x 256) filled via global_load_async_to_lds_b128
// (ASYNCcnt), two fills in flight.
// Fixed-shift logsumexp: |dot| <= 1 guarantees logits <= 1/T, and the
// diagonal term bounds each row sum >= exp(-2/T) >> 2^-126, so raw
// v_exp_f32 (flush-to-zero) is exact -> __builtin_amdgcn_exp2f.
// ---------------------------------------------------------------------------
__global__ __launch_bounds__(128, 1) void gemm_lse(const _Float16* __restrict__ qh,
                                                   const _Float16* __restrict__ kh,
                                                   const float* __restrict__ diag,
                                                   float* __restrict__ row_out,
                                                   int N) {
    __shared__ _Float16 lds[3][16 * LDSROW];

    const int t    = threadIdx.x;
    const int wave = t >> 5;
    const int ln   = t & 31;
    const int mrow = ln & 15;      // row (A) or col (B) within tile
    const int kh2  = ln >> 4;      // K-half selector per ISA fragment layout
    const int m0   = blockIdx.x * 128 + wave * 32;

    // low 32 bits of a generic LDS pointer == wave-relative DS address
    const unsigned lds_base = (unsigned)(size_t)(&lds[0][0]);

    // staging map: thread t owns one contiguous 64-byte run of one tile row,
    // contiguous on BOTH the global and LDS side -> 4 async b128 fills.
    const int sc  = t >> 3;                 // tile column 0..15
    const int skk = (t & 7) * 32;           // half offset within row
    const unsigned g_stage0   = (unsigned)(sc * DDIM + skk) * 2u;   // global byte off
    const unsigned l_stageoff = (unsigned)(sc * LDSROW + skk) * 2u; // lds byte off

#define STAGE_TILE(bufsel, jtile)                                              \
    do {                                                                       \
        unsigned goff = g_stage0 + (unsigned)(jtile) * (16u * DDIM * 2u);      \
        unsigned loff = lds_base + (unsigned)(bufsel) * TILEB + l_stageoff;    \
        asm volatile(                                                          \
            "global_load_async_to_lds_b128 %0, %1, %2 offset:0\n\t"            \
            "global_load_async_to_lds_b128 %0, %1, %2 offset:16\n\t"           \
            "global_load_async_to_lds_b128 %0, %1, %2 offset:32\n\t"           \
            "global_load_async_to_lds_b128 %0, %1, %2 offset:48"               \
            :: "v"(loff), "v"(goff), "s"(kh) : "memory");                      \
    } while (0)

    // ---- preload resident A fragments (ISA 16-bit A layout: lanes 0-15 hold
    //      K = 8*kh2..+7 and 16+8*kh2..+7 per 16x32 fragment) ----
    v16h a0[KSTEP], a1[KSTEP];
    {
        const _Float16* ab0 = qh + (size_t)(m0 + mrow) * DDIM + kh2 * 8;
        const _Float16* ab1 = qh + (size_t)(m0 + 16 + mrow) * DDIM + kh2 * 8;
#pragma unroll
        for (int ks = 0; ks < KSTEP; ++ks) {
            v8h lo0 = *(const v8h*)(ab0 + ks * 32);
            v8h hi0 = *(const v8h*)(ab0 + ks * 32 + 16);
            a0[ks] = __builtin_shufflevector(lo0, hi0, 0, 1, 2, 3, 4, 5, 6, 7,
                                             8, 9, 10, 11, 12, 13, 14, 15);
            v8h lo1 = *(const v8h*)(ab1 + ks * 32);
            v8h hi1 = *(const v8h*)(ab1 + ks * 32 + 16);
            a1[ks] = __builtin_shufflevector(lo1, hi1, 0, 1, 2, 3, 4, 5, 6, 7,
                                             8, 9, 10, 11, 12, 13, 14, 15);
        }
    }

    float s0[8], s1[8];
#pragma unroll
    for (int v = 0; v < 8; ++v) { s0[v] = 0.0f; s1[v] = 0.0f; }

    const int jtiles = N / 16;

    // prologue: fill buffers 0 and 1, wait for both (one-time serialization)
    STAGE_TILE(0, 0);
    if (jtiles > 1) STAGE_TILE(1, 1);
    asm volatile("s_wait_asynccnt 0x0" ::: "memory");
    __syncthreads();

    int cb = 0;   // buffer holding tile jt
    int fb = 2;   // buffer to fill with tile jt+2
    for (int jt = 0; jt < jtiles; ++jt) {
        const bool fill = (jt + 2 < jtiles);
        if (fill) STAGE_TILE(fb, jt + 2);

        // two 16x16 logits tiles: each B fragment feeds two independent WMMAs
        v8f acc0 = {};
        v8f acc1 = {};
        const _Float16* bb = &lds[cb][mrow * LDSROW + kh2 * 8];
#pragma unroll
        for (int ks = 0; ks < KSTEP; ++ks) {
            v8h lo = *(const v8h*)(bb + ks * 32);
            v8h hi = *(const v8h*)(bb + ks * 32 + 16);
            v16h b = __builtin_shufflevector(lo, hi, 0, 1, 2, 3, 4, 5, 6, 7,
                                             8, 9, 10, 11, 12, 13, 14, 15);
            acc0 = __builtin_amdgcn_wmma_f32_16x16x32_f16(
                false, a0[ks], false, b, (short)0, acc0, false, false);
            acc1 = __builtin_amdgcn_wmma_f32_16x16x32_f16(
                false, a1[ks], false, b, (short)0, acc1, false, false);
        }

        // sum of exp2((dot-1)*log2e/T): one fma + one raw v_exp_f32 + one add
#pragma unroll
        for (int v = 0; v < 8; ++v) {
            s0[v] += __builtin_amdgcn_exp2f(__builtin_fmaf(acc0[v], EXPSCL, -EXPSCL));
            s1[v] += __builtin_amdgcn_exp2f(__builtin_fmaf(acc1[v], EXPSCL, -EXPSCL));
        }

        // async loads complete in order: <=4 outstanding guarantees the
        // jt+1 fill has landed while the jt+2 fill stays in flight
        if (fill) asm volatile("s_wait_asynccnt 0x4" ::: "memory");
        else      asm volatile("s_wait_asynccnt 0x0" ::: "memory");
        __syncthreads();

        cb = (cb == 2) ? 0 : cb + 1;
        fb = (fb == 2) ? 0 : fb + 1;
    }
#undef STAGE_TILE

    // C/D layout: VGPR v, lanes 0-15 -> M=v, lanes 16-31 -> M=v+8; columns
    // stripe across the 16 lanes of each half -> butterfly over masks 1,2,4,8
#pragma unroll
    for (int v = 0; v < 8; ++v) {
        s0[v] += __shfl_xor(s0[v], 1, 32);
        s0[v] += __shfl_xor(s0[v], 2, 32);
        s0[v] += __shfl_xor(s0[v], 4, 32);
        s0[v] += __shfl_xor(s0[v], 8, 32);
        s1[v] += __shfl_xor(s1[v], 1, 32);
        s1[v] += __shfl_xor(s1[v], 2, 32);
        s1[v] += __shfl_xor(s1[v], 4, 32);
        s1[v] += __shfl_xor(s1[v], 8, 32);
    }
    if (mrow == 0) {
#pragma unroll
        for (int v = 0; v < 8; ++v) {
            const int row0 = m0 + kh2 * 8 + v;
            row_out[row0] = logf(s0[v]) + INV_T - diag[row0];
            const int row1 = m0 + 16 + kh2 * 8 + v;
            row_out[row1] = logf(s1[v]) + INV_T - diag[row1];
        }
    }
}

// ---------------------------------------------------------------------------
// Kernel 3: deterministic mean over rows -> scalar.
// ---------------------------------------------------------------------------
__global__ __launch_bounds__(256) void reduce_mean(const float* __restrict__ r,
                                                   float* __restrict__ out, int n) {
    __shared__ float sm[256];
    float acc = 0.0f;
    for (int i = threadIdx.x; i < n; i += 256) acc += r[i];
    sm[threadIdx.x] = acc;
    __syncthreads();
    for (int off = 128; off > 0; off >>= 1) {
        if ((int)threadIdx.x < off) sm[threadIdx.x] += sm[threadIdx.x + off];
        __syncthreads();
    }
    if (threadIdx.x == 0) out[0] = sm[0] / (float)n;
}

// ---------------------------------------------------------------------------
extern "C" void kernel_launch(void* const* d_in, const int* in_sizes, int n_in,
                              void* d_out, int out_size, void* d_ws, size_t ws_size,
                              hipStream_t stream) {
    const float* q = (const float*)d_in[0];
    const float* k = (const float*)d_in[1];
    const int N = in_sizes[0] / DDIM;   // 16384

    char* ws = (char*)d_ws;
    const size_t halfBytes = (size_t)N * DDIM * sizeof(_Float16);  // 8 MB
    _Float16* qh      = (_Float16*)ws;
    _Float16* kh      = (_Float16*)(ws + halfBytes);
    float*    diag    = (float*)(ws + 2 * halfBytes);
    float*    row_out = (float*)(ws + 2 * halfBytes + (size_t)N * sizeof(float));
    float*    out     = (float*)d_out;

    nrm_diag<<<N / 8, 256, 0, stream>>>(q, k, qh, kh, diag);
    gemm_lse<<<N / 128, 128, 0, stream>>>(qh, kh, diag, row_out, N);
    reduce_mean<<<1, 256, 0, stream>>>(row_out, out, N);
}
